// Decoder_86165633892728
// MI455X (gfx1250) — matrix operand order repaired and checked by reading, not compiled, not run on hardware
//
#include <hip/hip_runtime.h>
#include <cstddef>

// ======================================================================
// Decoder (M=8, K=256, D=256, H=8, DFF=1024, B=64, num_samples=16)
//  * seq-len-1 self-attention == identity on V  (q1/k1 dead)
//  * mask*(-1e9) => cross-attention restricted to 256-key segment i
//  * this_p = head-0 logits of segment i
//  * k2/v2 never materialized: logits via R = k2_w_h @ q2_h,
//    values via (aw @ enc_seg) @ v2_w
//  * JAX threefry2x32 gumbel sampling reproduced bit-exactly
// WMMA: v_wmma_f32_16x16x32_bf16 (logits GEMM), v_wmma_f32_16x16x4_f32
// (softmax-weighted enc row-sum). LDS B-fragments fetched as b128 pairs
// from a bank-conflict-free padded layout. Async-to-LDS staging in K4.
// ======================================================================

#define NB   64      // batch
#define ND   256     // model dim
#define NH   8       // heads
#define NKSEG 256    // codes per segment
#define NS   16      // num_samples
#define NDFF 1024
#define RLP  264     // padded LDS row stride (bf16) -> bank start 4*m mod 64

typedef __attribute__((ext_vector_type(16))) __bf16 v16bf_t;
typedef __attribute__((ext_vector_type(8)))  __bf16 v8bf_t;
typedef __attribute__((ext_vector_type(8)))  float  v8f_t;
typedef __attribute__((ext_vector_type(2)))  float  v2f_t;
typedef __attribute__((ext_vector_type(4)))  int    v4i_t;

__host__ __device__ inline unsigned rotl32(unsigned v, int r){ return (v<<r)|(v>>(32-r)); }

// JAX threefry2x32 block cipher (20 rounds, 5 key injections)
__host__ __device__ inline void threefry2x32(unsigned k0, unsigned k1,
                                             unsigned c0, unsigned c1,
                                             unsigned& o0, unsigned& o1){
  unsigned ks2 = k0 ^ k1 ^ 0x1BD11BDAu;
  unsigned x0 = c0 + k0, x1 = c1 + k1;
#define TF_MIX(r) { x0 += x1; x1 = rotl32(x1,(r)); x1 ^= x0; }
#define TF_RA TF_MIX(13) TF_MIX(15) TF_MIX(26) TF_MIX(6)
#define TF_RB TF_MIX(17) TF_MIX(29) TF_MIX(16) TF_MIX(24)
  TF_RA x0 += k1;  x1 += ks2 + 1u;
  TF_RB x0 += ks2; x1 += k0  + 2u;
  TF_RA x0 += k0;  x1 += k1  + 3u;
  TF_RB x0 += k1;  x1 += ks2 + 4u;
  TF_RA x0 += ks2; x1 += k0  + 5u;
#undef TF_RA
#undef TF_RB
#undef TF_MIX
  o0 = x0; o1 = x1;
}

__device__ inline float block_sum256(float v, float* red){
  int t = threadIdx.x;
  red[t] = v; __syncthreads();
  for(int s = 128; s > 0; s >>= 1){ if(t < s) red[t] += red[t+s]; __syncthreads(); }
  float r = red[0]; __syncthreads();
  return r;
}

// ---------------------------------------------------------------------
// K1: ctx -> v1 -> out1 -> q2 -> R (q2 folded through k2_w), lg bias.
// ---------------------------------------------------------------------
__global__ __launch_bounds__(256) void k1_prep(
    const float* __restrict__ x, const float* __restrict__ prev,
    const float* __restrict__ cum, int first,
    const float* __restrict__ v1w, const float* __restrict__ v1b,
    const float* __restrict__ o1w, const float* __restrict__ o1b,
    const float* __restrict__ q2w, const float* __restrict__ q2b,
    const float* __restrict__ k2w, const float* __restrict__ k2b,
    float* __restrict__ out1_ws, float* __restrict__ R_ws,
    float* __restrict__ lgb_ws)
{
  __shared__ float ctx[3*ND], v1s[ND], o1s[ND], q2s[ND];
  const int b = blockIdx.x, t = threadIdx.x;
  ctx[t]        = x[b*ND + t];
  ctx[ND + t]   = prev[b*ND + t];
  ctx[2*ND + t] = first ? 0.f : cum[b*ND + t];
  __syncthreads();
  float a = v1b[t];
  for(int d = 0; d < 3*ND; ++d) a += ctx[d] * v1w[d*ND + t];
  v1s[t] = a; __syncthreads();
  a = o1b[t];
  for(int d = 0; d < ND; ++d) a += v1s[d] * o1w[d*ND + t];
  o1s[t] = a; out1_ws[b*ND + t] = a; __syncthreads();
  a = q2b[t];
  for(int d = 0; d < ND; ++d) a += o1s[d] * q2w[d*ND + t];
  q2s[t] = a; __syncthreads();
  const float scale = 0.17677669529663687f; // 1/sqrt(D/H) = 1/sqrt(32)
  for(int h = 0; h < NH; ++h){
    float r = 0.f;
    for(int dd = 0; dd < 32; ++dd) r += k2w[t*ND + h*32 + dd] * q2s[h*32 + dd];
    R_ws[(b*NH + h)*ND + t] = r * scale;
  }
  if(t < NH){
    float r = 0.f;
    for(int dd = 0; dd < 32; ++dd) r += k2b[t*32 + dd] * q2s[t*32 + dd];
    lgb_ws[b*NH + t] = r * scale;
  }
}

// ---------------------------------------------------------------------
// K2: lg[b,h,k] = enc_seg_b(256x256) @ R_b^T(256x16-pad) via
// v_wmma_f32_16x16x32_bf16. R kept head-major in LDS with 264-bf16 row
// stride: each lane's B-fragment = 2x ds_load_b128, conflict-free.
// ---------------------------------------------------------------------
__global__ __launch_bounds__(256) void k2_logits_wmma(
    const float* __restrict__ enc, int seg,
    const float* __restrict__ R_ws, const float* __restrict__ lgb_ws,
    float* __restrict__ lg_ws)
{
  __shared__ __bf16 Rl[16*RLP];    // [head][k], heads 8..15 zero-padded
  __shared__ float  lgb[16];
  const int b = blockIdx.x, t = threadIdx.x;
  for(int n = 0; n < NH; ++n)  Rl[n*RLP + t] = (__bf16)R_ws[(b*NH + n)*ND + t];
  for(int n = NH; n < 16; ++n) Rl[n*RLP + t] = (__bf16)0.f;
  if(t < NH) lgb[t] = lgb_ws[b*NH + t];
  else if(t < 16) lgb[t] = 0.f;
  __syncthreads();

  const int wave = t >> 5, lane = t & 31;
  const int m = lane & 15, hi = lane >> 4;
  for(int ti = 0; ti < 2; ++ti){
    const int rowBase = wave*32 + ti*16;
    const float* arow = enc + ((size_t)(b*2048) + (size_t)seg + (size_t)(rowBase + m)) * ND;
    __builtin_prefetch(arow, 0, 0);
    v8f_t acc = {};
    for(int kb = 0; kb < ND; kb += 32){
      const int kA = hi * 8;
      float4 f0 = *(const float4*)(arow + kb + kA);
      float4 f1 = *(const float4*)(arow + kb + kA + 4);
      float4 f2 = *(const float4*)(arow + kb + 16 + kA);
      float4 f3 = *(const float4*)(arow + kb + 16 + kA + 4);
      v16bf_t A;
      A[0]=(__bf16)f0.x;  A[1]=(__bf16)f0.y;  A[2]=(__bf16)f0.z;  A[3]=(__bf16)f0.w;
      A[4]=(__bf16)f1.x;  A[5]=(__bf16)f1.y;  A[6]=(__bf16)f1.z;  A[7]=(__bf16)f1.w;
      A[8]=(__bf16)f2.x;  A[9]=(__bf16)f2.y;  A[10]=(__bf16)f2.z; A[11]=(__bf16)f2.w;
      A[12]=(__bf16)f3.x; A[13]=(__bf16)f3.y; A[14]=(__bf16)f3.z; A[15]=(__bf16)f3.w;
      // B fragment: lane needs Rl[head=m][kB .. kB+7] and [kB+16 .. kB+23] (16B aligned)
      const int kB = kb + hi*8;
      const v8bf_t blo = *(const v8bf_t*)(&Rl[m*RLP + kB]);
      const v8bf_t bhi = *(const v8bf_t*)(&Rl[m*RLP + kB + 16]);
      v16bf_t Bf = __builtin_shufflevector(blo, bhi,
                     0,1,2,3,4,5,6,7,8,9,10,11,12,13,14,15);
      acc = __builtin_amdgcn_wmma_f32_16x16x32_bf16(
              false, A, false, Bf, (short)0, acc, false, false);
    }
    if(m < NH){  // C tile: M=k-row, N=head (lane%16)
      #pragma unroll
      for(int j = 0; j < 8; ++j){
        const int kout = rowBase + j + hi*8;
        lg_ws[(b*NH + m)*NKSEG + kout] = acc[j] + lgb[m];
      }
    }
  }
}

// ---------------------------------------------------------------------
// K3: per-head softmax over 256 keys, then s[b,h,:] = aw @ enc_seg_b via
// v_wmma_f32_16x16x4_f32 (heads padded to 16). One block per b.
// ---------------------------------------------------------------------
__global__ __launch_bounds__(256) void k3_attn_s(
    const float* __restrict__ enc, int seg,
    const float* __restrict__ lg_ws, float* __restrict__ s_ws)
{
  __shared__ float aw[16*NKSEG];   // [head][k], heads 8..15 zero
  const int b = blockIdx.x, t = threadIdx.x;
  const int wave = t >> 5, lane = t & 31;
  { // wave h computes softmax of head h
    const int h = wave;
    float vals[8]; float mx = -1e30f;
    for(int u = 0; u < 8; ++u){
      float v = lg_ws[(b*NH + h)*NKSEG + lane + 32*u];
      vals[u] = v; mx = fmaxf(mx, v);
    }
    for(int off = 16; off; off >>= 1) mx = fmaxf(mx, __shfl_xor(mx, off, 32));
    float sm = 0.f;
    for(int u = 0; u < 8; ++u){ vals[u] = expf(vals[u] - mx); sm += vals[u]; }
    for(int off = 16; off; off >>= 1) sm += __shfl_xor(sm, off, 32);
    const float inv = 1.f / sm;
    for(int u = 0; u < 8; ++u) aw[h*NKSEG + lane + 32*u] = vals[u] * inv;
  }
  for(int r = NH; r < 16; ++r) aw[r*NKSEG + t] = 0.f;
  __syncthreads();

#if __has_builtin(__builtin_amdgcn_wmma_f32_16x16x4_f32)
  const int m = lane & 15, hi = lane >> 4;
  for(int nt = 0; nt < 2; ++nt){
    const int n0 = (wave*2 + nt)*16;
    v8f_t acc = {};
    for(int kb = 0; kb < NKSEG; kb += 4){
      const int ks = hi*2;
      v2f_t a; a.x = aw[m*NKSEG + kb + ks]; a.y = aw[m*NKSEG + kb + ks + 1];
      const float* bp = enc + ((size_t)(b*2048) + (size_t)seg + (size_t)(kb + ks)) * ND + n0 + m;
      v2f_t bb; bb.x = bp[0]; bb.y = bp[ND];
      acc = __builtin_amdgcn_wmma_f32_16x16x4_f32(
              false, a, false, bb, (short)0, acc, false, false);
    }
    #pragma unroll
    for(int j = 0; j < 8; ++j){
      const int h = j + hi*8;     // C rows = heads; 8..15 are padding
      if(h < NH) s_ws[(b*NH + h)*ND + n0 + m] = acc[j];
    }
  }
#else
  float sacc[NH];
  for(int h = 0; h < NH; ++h) sacc[h] = 0.f;
  for(int k = 0; k < NKSEG; ++k){
    float ev = enc[((size_t)(b*2048) + (size_t)seg + (size_t)k) * ND + t];
    for(int h = 0; h < NH; ++h) sacc[h] += aw[h*NKSEG + k] * ev;
  }
  for(int h = 0; h < NH; ++h) s_ws[(b*NH + h)*ND + t] = sacc[h];
#endif
}

// ---------------------------------------------------------------------
// K4: attn2 = (s @ v2_w + v2_b) @ o2_w + o2_b ; out2 = LN(attn2+out1);
//     ffn -> out3 = LN(ffn+out2). One block per b; 256 threads.
// s tile (8KB) staged into LDS with async DMA (ASYNCcnt-tracked).
// ---------------------------------------------------------------------
__global__ __launch_bounds__(256) void k4_ffn(
    const float* __restrict__ out1_ws, const float* __restrict__ s_ws,
    const float* __restrict__ v2w, const float* __restrict__ v2b,
    const float* __restrict__ o2w, const float* __restrict__ o2b,
    const float* __restrict__ g2, const float* __restrict__ b2,
    const float* __restrict__ g3, const float* __restrict__ b3,
    const float* __restrict__ f1w, const float* __restrict__ f1b,
    const float* __restrict__ f2w, const float* __restrict__ f2b,
    float* __restrict__ out3_ws)
{
  __shared__ float sl[NH*ND], am[ND], o2l[ND], hid[NDFF], red[ND];
  const int b = blockIdx.x, t = threadIdx.x;
#if __has_builtin(__builtin_amdgcn_global_load_async_to_lds_b128) && \
    __has_builtin(__builtin_amdgcn_s_wait_asynccnt)
  {
    // each thread DMAs 32 bytes global->LDS (ASYNCcnt-tracked).
    // param0: v4i* in global (__device__) AS per compiler diagnostic;
    // AS1 value == flat address. AS3 value == low 32 bits of flat LDS addr.
    const float* g = s_ws + (size_t)(b*NH)*ND + t*8;
    float* l = &sl[t*8];
    __attribute__((address_space(1))) v4i_t* g0 =
        (__attribute__((address_space(1))) v4i_t*)(unsigned long long)(size_t)g;
    __attribute__((address_space(3))) v4i_t* l0 =
        (__attribute__((address_space(3))) v4i_t*)(unsigned)(size_t)l;
    __builtin_amdgcn_global_load_async_to_lds_b128(g0,     l0,     0, 0);
    __builtin_amdgcn_global_load_async_to_lds_b128(g0 + 1, l0 + 1, 0, 0);
    __builtin_amdgcn_s_wait_asynccnt(0);
  }
  __syncthreads();
#else
  for(int h = 0; h < NH; ++h) sl[h*ND + t] = s_ws[(b*NH + h)*ND + t];
  __syncthreads();
#endif
  const int h = t >> 5;                 // head of output column t
  float a = v2b[t];                     // sum_k aw = 1 => +v2_b exact
  for(int d = 0; d < ND; ++d) a += sl[h*ND + d] * v2w[d*ND + t];
  am[t] = a; __syncthreads();
  a = o2b[t];
  for(int d = 0; d < ND; ++d) a += am[d] * o2w[d*ND + t];
  const float r = a + out1_ws[b*ND + t];
  float mean = block_sum256(r, red) * (1.f/ND);
  float dv = r - mean;
  float var = block_sum256(dv*dv, red) * (1.f/ND);
  float o2v = dv * rsqrtf(var + 1e-6f) * g2[t] + b2[t];
  o2l[t] = o2v; __syncthreads();
  for(int j = 0; j < 4; ++j){
    const int hh = t + 256*j;
    float hv = f1b[hh];
    for(int d = 0; d < ND; ++d) hv += o2l[d] * f1w[d*NDFF + hh];
    hid[hh] = fmaxf(hv, 0.f);
  }
  __syncthreads();
  float fo = f2b[t];
  for(int d = 0; d < NDFF; ++d) fo += hid[d] * f2w[d*ND + t];
  const float r2 = fo + o2l[t];
  float mean2 = block_sum256(r2, red) * (1.f/ND);
  float dv2 = r2 - mean2;
  float var2 = block_sum256(dv2*dv2, red) * (1.f/ND);
  out3_ws[b*ND + t] = dv2 * rsqrtf(var2 + 1e-6f) * g3[t] + b3[t];
}

// ---------------------------------------------------------------------
// K5: gumbel-argmax sampling (JAX threefry-exact) + log picked prob.
// ---------------------------------------------------------------------
__global__ __launch_bounds__(256) void k5_sample(
    const float* __restrict__ lg_ws, unsigned fk0, unsigned fk1,
    int layer, float* __restrict__ pi_out, int* __restrict__ idx_ws)
{
  __shared__ float tp[NKSEG], red[256];
  const int b = blockIdx.x, t = threadIdx.x;
  tp[t] = lg_ws[(b*NH + 0)*NKSEG + t];   // head-0 logits = this_p
  __syncthreads();
  red[t] = tp[t]; __syncthreads();
  for(int s = 128; s; s >>= 1){ if(t < s) red[t] = fmaxf(red[t], red[t+s]); __syncthreads(); }
  const float mx = red[0]; __syncthreads();
  red[t] = expf(tp[t] - mx); __syncthreads();
  for(int s = 128; s; s >>= 1){ if(t < s) red[t] += red[t+s]; __syncthreads(); }
  const float lse = mx + logf(red[0]); __syncthreads();

  const int wave = t >> 5, lane = t & 31;
  for(int si = 0; si < 2; ++si){
    const int s = wave + 8*si;
    float bestv = -1e38f; int besti = 0;
    for(int u8 = 0; u8 < 8; ++u8){
      const int k = lane + 32*u8;
      const unsigned j = (unsigned)((b*NS + s)*NKSEG + k);
      const unsigned p = j & 0x1FFFFu, w = j >> 17; // split at 131072
      unsigned o0, o1;
      threefry2x32(fk0, fk1, p, p + 0x20000u, o0, o1);
      const unsigned bits = w ? o1 : o0;
      float f = __uint_as_float((bits >> 9) | 0x3f800000u) - 1.0f;
      float u = f * (1.0f - 1e-7f) + 1e-7f;
      u = fmaxf(u, 1e-7f);
      const float g = tp[k] - logf(-logf(u));
      if(g > bestv){ bestv = g; besti = k; }
    }
    for(int off = 16; off; off >>= 1){
      float ov = __shfl_xor(bestv, off, 32);
      int   oi = __shfl_xor(besti, off, 32);
      if(ov > bestv || (ov == bestv && oi < besti)){ bestv = ov; besti = oi; }
    }
    if(lane == 0){
      idx_ws[b*NS + s] = besti;
      pi_out[(layer*NB + b)*NS + s] = tp[besti] - lse;
    }
  }
}

// ---------------------------------------------------------------------
// K6: codebook gather; quantized accumulate; cumulative += mean(samples)
// ---------------------------------------------------------------------
__global__ __launch_bounds__(256) void k6_gather(
    const float* __restrict__ codebook, int seg,
    const int* __restrict__ idx_ws, int first,
    float* __restrict__ quant_out, float* __restrict__ cum_ws)
{
  const int b = blockIdx.x, t = threadIdx.x;
  float csum = 0.f;
  for(int s = 0; s < NS; ++s){
    const int idx = idx_ws[b*NS + s];
    const float v = codebook[((size_t)(b*2048) + (size_t)seg + (size_t)idx) * ND + t];
    float* q = quant_out + ((size_t)(b*NS + s)) * ND + t;
    if(first) *q = v; else *q += v;
    csum += v;
  }
  float* c = cum_ws + b*ND + t;
  if(first) *c = csum * (1.f/NS); else *c += csum * (1.f/NS);
}

// ======================================================================
// Host side
// ======================================================================
extern "C" void kernel_launch(void* const* d_in, const int* in_sizes, int n_in,
                              void* d_out, int out_size, void* d_ws, size_t ws_size,
                              hipStream_t stream)
{
  (void)in_sizes; (void)n_in; (void)out_size; (void)ws_size;
  const float* x        = (const float*)d_in[0];
  const float* codebook = (const float*)d_in[1];
  const float* enc      = (const float*)d_in[2];
  const float* initv    = (const float*)d_in[3];

  // params flattened pytree-style: layers in order, dict keys sorted.
  // non-last (24): f1_b,f1_w,f2_b,f2_w,k1_b,k1_w,k2_b,k2_w,ln2_b,ln2_g,
  //   ln3_b,ln3_g,o1_b,o1_w,o2_b,o2_w,q1_b,q1_w,q2_b,q2_w,v1_b,v1_w,v2_b,v2_w
  // last (12): k1_b,k1_w,k2_b,k2_w,o1_b,o1_w,q1_b,q1_w,q2_b,q2_w,v1_b,v1_w

  float* ws   = (float*)d_ws;
  float* out1 = ws;               // 16384
  float* out3 = ws + 16384;       // 16384
  float* cum  = ws + 32768;       // 16384
  float* R    = ws + 49152;       // 131072
  float* lgb  = ws + 180224;      // 512
  float* lg   = ws + 180736;      // 131072
  float* sws  = ws + 311808;      // 131072
  int*   idx  = (int*)(ws + 442880); // 1024 ints

  float* pi    = (float*)d_out;          // (8,64,16)
  float* quant = (float*)d_out + 8192;   // (64,16,256)

  for(int i = 0; i < 8; ++i){
    const bool last = (i == 7);
    const int base = last ? (4 + 24*7) : (4 + 24*i);
    auto p = [&](int si){ return (const float*)d_in[base + si]; };
    const float *v1w, *v1b, *o1w, *o1b, *q2w, *q2b, *k2w, *k2b;
    if(!last){ v1w=p(21); v1b=p(20); o1w=p(13); o1b=p(12);
               q2w=p(19); q2b=p(18); k2w=p(7);  k2b=p(6); }
    else     { v1w=p(11); v1b=p(10); o1w=p(5);  o1b=p(4);
               q2w=p(9);  q2b=p(8);  k2w=p(3);  k2b=p(2); }

    const float* prev = (i == 0) ? initv : out3;
    k1_prep<<<NB, 256, 0, stream>>>(x, prev, cum, i == 0 ? 1 : 0,
                                    v1w, v1b, o1w, o1b, q2w, q2b, k2w, k2b,
                                    out1, R, lgb);
    k2_logits_wmma<<<NB, 256, 0, stream>>>(enc, i*NKSEG, R, lgb, lg);
    if(!last){
      k3_attn_s<<<NB, 256, 0, stream>>>(enc, i*NKSEG, lg, sws);
      k4_ffn<<<NB, 256, 0, stream>>>(out1, sws,
                                     p(23), p(22),   // v2_w, v2_b
                                     p(15), p(14),   // o2_w, o2_b
                                     p(9),  p(8),    // ln2_g, ln2_b
                                     p(11), p(10),   // ln3_g, ln3_b
                                     p(1),  p(0),    // f1_w, f1_b
                                     p(3),  p(2),    // f2_w, f2_b
                                     out3);
    }
    // folded key = threefry2x32(key=[0,42], count=[0,i])
    unsigned fk0, fk1;
    threefry2x32(0u, 42u, 0u, (unsigned)i, fk0, fk1);
    k5_sample<<<NB, 256, 0, stream>>>(lg, fk0, fk1, i, pi, idx);
    k6_gather<<<NB, 256, 0, stream>>>(codebook, i*NKSEG, idx,
                                      i == 0 ? 1 : 0, quant, cum);
  }
}
